// ModelSimplest_27625229648359
// MI455X (gfx1250) — matrix operand order repaired and checked by reading, not compiled, not run on hardware
//
#include <hip/hip_runtime.h>
#include <hip/hip_bf16.h>
#include <math.h>

typedef __bf16 bf16_t;
typedef __attribute__((ext_vector_type(16))) __bf16 v16bf;
typedef __attribute__((ext_vector_type(8)))  float   v8f;
typedef __attribute__((ext_vector_type(4)))  float   v4f;

#define S_IN   18
#define KW     13
#define S_OUT  6
#define X_BSTR (S_IN*S_IN*S_IN*S_IN)   // 104976
#define X_S1   (S_IN*S_IN*S_IN)        // 5832
#define X_S2   (S_IN*S_IN)             // 324
#define SLAB   (S_IN*S_IN)             // 324 = (j3,j4) window, contiguous
#define SLABP  352                     // padded to 11 * 32
#define NK12   (KW*KW)                 // 169 (k1,k2) slabs
#define KPITCH (NK12*SLABP)            // 59488 bf16 per B column
#define NTILES 7
#define NPAD   (NTILES*16)             // 112
#define NUSED  108                     // 3 * 6 * 6 = (c,o3,o4)
#define KCHUNKS (SLABP/32)             // 11
#define MROWS  (1024*S_OUT*S_OUT)      // 36864 = (b,o1,o2)
#define BATCH  1024
#define ASYNC_PER_SLAB 41              // ceil(16*81 / 32) async b128 issues/wave

#if defined(__has_builtin)
#if __has_builtin(__builtin_amdgcn_global_load_async_to_lds_b128)
#define HAS_ASYNC_LDS 1
#endif
#endif

#ifdef HAS_ASYNC_LDS
// Parameter types per hipcc diagnostic: AS1 / AS3 pointers to int4 vectors.
typedef int async_v4i __attribute__((vector_size(4 * sizeof(int))));
typedef __attribute__((address_space(1))) async_v4i* async_gptr;
typedef __attribute__((address_space(3))) async_v4i* async_lptr;
#define WAIT_ASYNC(n) asm volatile("s_wait_asynccnt %0" :: "i"(n) : "memory")
#else
#define WAIT_ASYNC(n) do {} while (0)
#endif
#define WAIT_DS0() asm volatile("s_wait_dscnt 0" ::: "memory")

// ---------------------------------------------------------------------------
// Expand W4 (3,13,13,13,13) into windowed-GEMM B matrix, bf16, n-major:
//   Bexp[n][k12*352 + (j3*18+j4)] = W4[c,k1,k2,j3-o3,j4-o4]  (0 outside)
//   n = c*36 + o3*6 + o4  (108 used, 4 zero-pad columns)
// ---------------------------------------------------------------------------
__global__ void expand_weights_kernel(const float* __restrict__ W4,
                                      bf16_t* __restrict__ Bexp, int total) {
  for (int idx = blockIdx.x * blockDim.x + threadIdx.x; idx < total;
       idx += gridDim.x * blockDim.x) {
    const int n  = idx / KPITCH;
    const int q  = idx - n * KPITCH;
    const int k12 = q / SLABP;
    const int ql  = q - k12 * SLABP;
    float val = 0.0f;
    if (n < NUSED && ql < SLAB) {
      const int j3 = ql / S_IN, j4 = ql - j3 * S_IN;
      const int c  = n / 36;
      const int nr = n - c * 36;
      const int o3 = nr / 6, o4 = nr - o3 * 6;
      const int d3 = j3 - o3, d4 = j4 - o4;
      if (d3 >= 0 && d3 < KW && d4 >= 0 && d4 < KW) {
        const int k1 = k12 / KW, k2 = k12 - k1 * KW;
        val = W4[((c * KW + k1) * KW + k2) * (KW * KW) + d3 * KW + d4];
      }
    }
    Bexp[idx] = (bf16_t)val;
  }
}

__global__ void zero_kernel(float* __restrict__ p, int n) {
  const int i = blockIdx.x * blockDim.x + threadIdx.x;
  if (i < n) p[i] = 0.0f;
}

__global__ void finalize_kernel(const float* __restrict__ accB,
                                const float* __restrict__ blin,
                                float* __restrict__ out, int n) {
  const int i = blockIdx.x * blockDim.x + threadIdx.x;
  if (i < n) out[i] = 1.0f / (1.0f + __expf(-(accB[i] + blin[0])));
}

// ---------------------------------------------------------------------------
// Stage one (k1,k2) A-slab: 16 rows x 324 contiguous f32, async global->LDS.
// ---------------------------------------------------------------------------
__device__ __forceinline__ void stage_slab(const float* __restrict__ x,
                                           float (*sh)[16][SLABP], int buf,
                                           int k12, int tileBase, int lane) {
  const int k1 = k12 / KW, k2 = k12 - k1 * KW;
  for (int t = lane; t < 16 * (SLAB / 4); t += 32) {  // 16 rows * 81 b128
    const int row = t / (SLAB / 4);
    const int ch  = t - row * (SLAB / 4);
    const int r   = tileBase + row;
    const int b   = r / 36;
    const int rem = r - b * 36;
    const int o1  = rem / 6;
    const int o2  = rem - o1 * 6;
    const float* gp = x + (size_t)b * X_BSTR + (size_t)(o1 + k1) * X_S1 +
                      (size_t)(o2 + k2) * X_S2 + (size_t)ch * 4;
    float* lp = &sh[buf][row][ch * 4];
#ifdef HAS_ASYNC_LDS
    __builtin_amdgcn_global_load_async_to_lds_b128(
        (async_gptr)gp, (async_lptr)lp, 0, 0);
#else
    *(v4f*)lp = *(const v4f*)gp;
#endif
  }
}

// ---------------------------------------------------------------------------
// Main fused kernel: windowed implicit-GEMM conv4d (bf16 WMMA, f32 acc),
// + bias + relu + dot(Wlin) reduction into per-batch accumulator.
// One wave per 16-row M-tile; double-buffered LDS A staging.
// ---------------------------------------------------------------------------
__global__ __launch_bounds__(32) void conv4d_wmma_kernel(
    const float* __restrict__ x, const bf16_t* __restrict__ Bexp,
    const float* __restrict__ b4, const float* __restrict__ Wlin,
    float* __restrict__ accB) {
  __shared__ __align__(128) float shA[2][16][SLABP];
  const int lane = threadIdx.x;
  const int half = lane >> 4;
  const int l16  = lane & 15;
  const int tileBase = blockIdx.x * 16;

  // Zero the K-pad region [324,352) once; staging never touches it.
  for (int t = lane; t < 2 * 16 * (SLABP - SLAB); t += 32) {
    const int br = t / (SLABP - SLAB);
    shA[br >> 4][br & 15][SLAB + (t % (SLABP - SLAB))] = 0.0f;
  }

  v8f acc[NTILES];
#pragma unroll
  for (int t = 0; t < NTILES; ++t)
#pragma unroll
    for (int i = 0; i < 8; ++i) acc[t][i] = 0.0f;

  WAIT_DS0();
  stage_slab(x, shA, 0, 0, tileBase, lane);

  for (int k12 = 0; k12 < NK12; ++k12) {
    const int buf = k12 & 1;
    if (k12 + 1 < NK12) {
      WAIT_DS0();  // prior iteration's reads of buf^1 have completed
      stage_slab(x, shA, buf ^ 1, k12 + 1, tileBase, lane);
      WAIT_ASYNC(ASYNC_PER_SLAB);  // current slab landed; next stays in flight
      __builtin_prefetch(Bexp + (size_t)l16 * KPITCH + (size_t)(k12 + 1) * SLABP,
                         0, 1);
    } else {
      WAIT_ASYNC(0);
    }

    const bf16_t* bbase = Bexp + (size_t)k12 * SLABP;

    for (int kc = 0; kc < KCHUNKS; ++kc) {
      // A fragment (16x32 bf16): lane {m=l16, half} holds K = half*16+[0,16)
      // -> two aligned v8f LDS reads + cvt to bf16.
      const float* ap = &shA[buf][l16][kc * 32 + half * 16];
      const v8f lo = *(const v8f*)ap;
      const v8f hi = *(const v8f*)(ap + 8);
      v16bf afrag;
#pragma unroll
      for (int i = 0; i < 8; ++i) {
        afrag[i]     = (bf16_t)lo[i];
        afrag[i + 8] = (bf16_t)hi[i];
      }
#pragma unroll
      for (int t = 0; t < NTILES; ++t) {
        // B fragment (32x16 bf16): lane {n=l16, half} holds 16 consecutive K
        // at fixed n -> one 32B load from n-major Bexp.
        const v16bf bfrag = *(const v16bf*)(
            bbase + (size_t)(t * 16 + l16) * KPITCH + kc * 32 + half * 16);
        acc[t] = __builtin_amdgcn_wmma_f32_16x16x32_bf16(
            false, afrag, false, bfrag, (short)0, acc[t], false, false);
      }
    }
  }

  // Epilogue: C layout -> lane holds col n=l16, rows M = half*8 + vv.
  // bias + relu + weight by Wlin, reduce across the 16 n-lanes, one
  // atomicAdd per output row into the per-batch accumulator.
#pragma unroll
  for (int vv = 0; vv < 8; ++vv) {
    const int r   = tileBase + half * 8 + vv;
    const int b   = r / 36;
    const int rem = r - b * 36;
    const int o1  = rem / 6;
    const int o2  = rem - o1 * 6;
    float p = 0.0f;
#pragma unroll
    for (int t = 0; t < NTILES; ++t) {
      const int n = t * 16 + l16;
      if (n < NUSED) {
        const int c  = n / 36;
        const int nr = n - c * 36;
        const int o3 = nr / 6;
        const int o4 = nr - o3 * 6;
        float v = acc[t][vv] + b4[c];
        v = v > 0.0f ? v : 0.0f;
        p += v * Wlin[c * 1296 + o1 * 216 + o2 * 36 + o3 * 6 + o4];
      }
    }
#pragma unroll
    for (int m = 8; m >= 1; m >>= 1) p += __shfl_xor(p, m, 16);
    if (l16 == 0) atomicAdd(&accB[b], p);
  }
}

extern "C" void kernel_launch(void* const* d_in, const int* in_sizes, int n_in,
                              void* d_out, int out_size, void* d_ws,
                              size_t ws_size, hipStream_t stream) {
  (void)in_sizes; (void)n_in; (void)out_size; (void)ws_size;
  const float* x    = (const float*)d_in[0];
  const float* W4   = (const float*)d_in[1];
  const float* b4   = (const float*)d_in[2];
  const float* Wlin = (const float*)d_in[3];
  const float* blin = (const float*)d_in[4];
  float*  out  = (float*)d_out;
  float*  accB = (float*)d_ws;                      // 1024 f32 accumulators
  bf16_t* Bexp = (bf16_t*)((char*)d_ws + 8192);     // 112 * 59488 bf16 ~13.3MB

  const int totalB = NPAD * KPITCH;
  expand_weights_kernel<<<4096, 256, 0, stream>>>(W4, Bexp, totalB);
  zero_kernel<<<(BATCH + 255) / 256, 256, 0, stream>>>(accB, BATCH);
  conv4d_wmma_kernel<<<MROWS / 16, 32, 0, stream>>>(x, Bexp, b4, Wlin, accB);
  finalize_kernel<<<(BATCH + 255) / 256, 256, 0, stream>>>(accB, blin, out,
                                                           BATCH);
}
